// MPNEncoder_24979529794282
// MI455X (gfx1250) — matrix operand order repaired and checked by reading, not compiled, not run on hardware
//
#include <hip/hip_runtime.h>

#define N_NODES 35000
#define N_MSGS  80000
#define MAX_NB  6
#define IN_DIM  128
#define HDIM    256

// bf16 row-major LDS tiles, padded (+8 bf16 / +4 dwords) for bank-conflict-free
// WMMA A-fragment reads (stride % 64 == 4 in dwords).
#define HN_STRIDE  132   // dwords per row: 256 bf16 + 8 pad   (528 B)
#define XA_STRIDE  196   // dwords per row: 384 bf16 + 8 pad   (784 B)
#define FM_STRIDE  68    // dwords per row: 128 bf16 + 8 pad

typedef __attribute__((ext_vector_type(16))) __bf16 v16bf;
typedef __attribute__((ext_vector_type(8)))  float  v8f;

union FragBF { unsigned u[8]; v16bf v; };

__device__ __forceinline__ unsigned f2bf(float x) {
  unsigned u = __float_as_uint(x);
  return (u + 0x7FFFu + ((u >> 16) & 1u)) >> 16;   // round-to-nearest-even
}
__device__ __forceinline__ unsigned pack2(float lo, float hi) {
  return f2bf(lo) | (f2bf(hi) << 16);
}
__device__ __forceinline__ float bf2f(unsigned bits16) {
  return __uint_as_float(bits16 << 16);
}
__device__ __forceinline__ v8f vzero8() {
  v8f z;
#pragma unroll
  for (int i = 0; i < 8; ++i) z[i] = 0.f;
  return z;
}
__device__ __forceinline__ v8f wmma_bf16(v16bf a, v16bf b, v8f c) {
  // D = A(16x32 bf16) * B(32x16 bf16) + C(f32)
  return __builtin_amdgcn_wmma_f32_16x16x32_bf16(false, a, false, b, (short)0, c,
                                                 false, false);
}
__device__ __forceinline__ float sigmoidf_(float x) { return 1.f / (1.f + __expf(-x)); }

// CDNA5 async memory->LDS copy: 16 bytes per lane, tracked by ASYNCcnt.
// LDS offset = low 32 bits of the generic pointer (aperture lives in the
// high 32 bits; HW adds LDS_BASE).
__device__ __forceinline__ void async_cp16(void* lds_ptr, const void* gptr) {
  unsigned ldsoff = (unsigned)(size_t)lds_ptr;
  asm volatile("global_load_async_to_lds_b128 %0, %1, off"
               :: "v"(ldsoff), "v"(gptr) : "memory");
}
__device__ __forceinline__ void wait_async0() {
  asm volatile("s_wait_asynccnt 0x0" ::: "memory");
}

// A fragment (16x32 bf16) from row-major bf16 LDS -> two ds_load_b128.
__device__ __forceinline__ v16bf load_a_frag(const unsigned* lds, int rowBase,
                                             int strideU32, int k0, int lane) {
  const int m = lane & 15, half = lane >> 4;
  const int base = (rowBase + m) * strideU32 + (k0 >> 1) + half * 4;
  FragBF f;
#pragma unroll
  for (int j = 0; j < 4; ++j) f.u[j] = lds[base + j];
#pragma unroll
  for (int j = 0; j < 4; ++j) f.u[4 + j] = lds[base + 8 + j];
  return f.v;
}

// B fragment (32x16 bf16) from pre-packed weights; per-lane words contiguous
// -> two coalesced global_load_b128 (1 KB sequential per fragment).
__device__ __forceinline__ v16bf load_b_frag(const unsigned* pk, int kt, int nt,
                                             int lane) {
  FragBF f;
  const unsigned* p = pk + ((size_t)(kt * 16 + nt) * 32 + lane) * 8;
#pragma unroll
  for (int j = 0; j < 8; ++j) f.u[j] = p[j];
  return f.v;
}

// ---- weight packer: f32 [K,256] row-major -> bf16 B-fragment order ----
__global__ __launch_bounds__(256)
void pack_weight_kernel(const float* __restrict__ W, unsigned* __restrict__ dst) {
  const int b = blockIdx.x;                 // b = kt*16 + nt
  const int t = threadIdx.x;
  const int lane = t & 31, j = t >> 5;
  const int kt = b >> 4;
  const int half = lane >> 4;
  const int n = ((b & 15) * 16) + (lane & 15);
  const int k = kt * 32 + (j >> 2) * 16 + half * 8 + (j & 3) * 2;   // B-operand K pattern
  const unsigned lo = f2bf(W[(size_t)k * HDIM + n]);
  const unsigned hi = f2bf(W[(size_t)(k + 1) * HDIM + n]);
  dst[((size_t)b * 32 + lane) * 8 + j] = lo | (hi << 16);
}

// ---- f32 -> packed bf16 pairs (for fmess, reused by all 5 steps) ----
__global__ __launch_bounds__(256)
void tobf16_kernel(const float* __restrict__ src, unsigned* __restrict__ dst,
                   int n_u32) {
  const int stride = gridDim.x * 256;
  for (int i = blockIdx.x * 256 + threadIdx.x; i < n_u32; i += stride)
    dst[i] = pack2(src[2 * i], src[2 * i + 1]);
}

// ---- rx = fmess @ W_r (loop-invariant) ----
__global__ __launch_bounds__(128)
void rx_kernel(const float* __restrict__ fmess, const unsigned* __restrict__ Wr_pk,
               float* __restrict__ rx) {
  __shared__ unsigned fm[16 * FM_STRIDE];
  const int t = threadIdx.x;
  const int e0 = blockIdx.x * 16;
  const int sub = t & 15, rg = t >> 4;
  for (int r = rg; r < 16; r += 8) {
    const float4* src = (const float4*)(fmess + (size_t)(e0 + r) * IN_DIM) + sub * 2;
    unsigned* dst = fm + r * FM_STRIDE + sub * 4;
#pragma unroll
    for (int i = 0; i < 2; ++i) {
      float4 x = src[i];
      dst[2 * i]     = pack2(x.x, x.y);
      dst[2 * i + 1] = pack2(x.z, x.w);
    }
  }
  __syncthreads();
  const int lane = t & 31, wave = t >> 5;
  const int mlo = lane & 15, half = lane >> 4;
  for (int nti = 0; nti < 4; ++nti) {
    const int nt = wave + nti * 4;
    const int n = nt * 16 + mlo;
    v8f acc = vzero8();
    for (int kt = 0; kt < 4; ++kt)
      acc = wmma_bf16(load_a_frag(fm, 0, FM_STRIDE, kt * 32, lane),
                      load_b_frag(Wr_pk, kt, nt, lane), acc);
#pragma unroll
    for (int v = 0; v < 8; ++v)
      rx[(size_t)(e0 + v + 8 * half) * HDIM + n] = acc[v];
  }
}

// ---- one fused GRU message-passing step; block = 16 edges, 4 waves ----
// h_in / fmess_bf are bf16; h_out is bf16 (intermediate) or f32 (final step).
template <bool OUT_F32>
__global__ __launch_bounds__(128)
void mpn_step_kernel(const unsigned short* __restrict__ h_in,
                     void* __restrict__ h_out,
                     const unsigned short* __restrict__ fmess_bf,
                     const int* __restrict__ bgraph,
                     const float* __restrict__ rx,
                     const unsigned* __restrict__ Wz_pk,
                     const unsigned* __restrict__ Ur_pk,
                     const unsigned* __restrict__ Wh_pk,
                     const float* __restrict__ b_z, const float* __restrict__ b_ur,
                     const float* __restrict__ b_h) {
  __shared__ unsigned hnei[96 * HN_STRIDE];   // rows: nb*16 + e  (bf16)
  __shared__ unsigned xa[16 * XA_STRIDE];     // [fmess | sum_h], later sum_h -> sum_gated
  __shared__ int bidx[96];
  const int t = threadIdx.x;
  const int e0 = blockIdx.x * 16;
  const int lane = t & 31, wave = t >> 5;

  if (t < 96) bidx[t] = bgraph[(size_t)(e0 + (t & 15)) * MAX_NB + (t >> 4)];
  __syncthreads();

  // Async DMA gathers straight into LDS (no VGPR round-trip, ASYNCcnt-tracked).
  // One h row (512 B bf16) per wave-instruction: 32 lanes x 16 B.
  for (int r = wave; r < 96; r += 4) {
    const char* g = (const char*)h_in + ((size_t)bidx[r] << 9) + lane * 16;
    async_cp16((char*)hnei + r * (HN_STRIDE * 4) + lane * 16, g);
  }
  // Two fmess rows (256 B bf16 each) per wave-instruction.
  for (int i = wave; i < 8; i += 4) {
    const int r = 2 * i + (lane >> 4);
    const char* g = (const char*)fmess_bf + ((size_t)(e0 + r) << 8) + (lane & 15) * 16;
    async_cp16((char*)xa + r * (XA_STRIDE * 4) + (lane & 15) * 16, g);
  }
  wait_async0();        // wave-local async completion
  __syncthreads();      // cross-wave visibility

  // sum_h -> xa rows 0..15, cols 128..383
  {
    const int e = t >> 3;            // 0..15
    const int w0 = (t & 7) * 16;     // dword offset within the 128-dword span
#pragma unroll
    for (int i = 0; i < 16; ++i) {
      float s0 = 0.f, s1 = 0.f;
#pragma unroll
      for (int nb = 0; nb < 6; ++nb) {
        const unsigned w = hnei[(nb * 16 + e) * HN_STRIDE + w0 + i];
        s0 += bf2f(w & 0xFFFFu);
        s1 += bf2f(w >> 16);
      }
      xa[e * XA_STRIDE + 64 + w0 + i] = pack2(s0, s1);
    }
  }
  __syncthreads();

  const int mlo = lane & 15, half = lane >> 4;

  // Phase 2: z = sigmoid([fmess|sum_h] @ W_z + b_z); snapshot z and sum_h in
  // registers (sum_h region gets overwritten by sum_gated below).
  float z_r[4][8], sh_r[4][8];
#pragma unroll
  for (int nti = 0; nti < 4; ++nti) {
    const int nt = wave + nti * 4;
    const int n = nt * 16 + mlo;
    v8f acc = vzero8();
    for (int kt = 0; kt < 12; ++kt)
      acc = wmma_bf16(load_a_frag(xa, 0, XA_STRIDE, kt * 32, lane),
                      load_b_frag(Wz_pk, kt, nt, lane), acc);
    const float bzv = b_z[n];
#pragma unroll
    for (int v = 0; v < 8; ++v) {
      const int M = v + 8 * half;
      z_r[nti][v] = sigmoidf_(acc[v] + bzv);
      const unsigned w = xa[M * XA_STRIDE + 64 + (n >> 1)];
      sh_r[nti][v] = bf2f((n & 1) ? (w >> 16) : (w & 0xFFFFu));
    }
  }
  __syncthreads();

  // Phase 3: GEMM2  h_nei @ U_r with fused r-gate + gated neighbor sum.
  // U_r B-fragments are nb-invariant: cache all 8 in registers, reuse 6x.
  for (int nti = 0; nti < 4; ++nti) {
    const int nt = wave + nti * 4;
    const int n = nt * 16 + mlo;
    const float bu = b_ur[n];
    v16bf bU[8];
#pragma unroll
    for (int kt = 0; kt < 8; ++kt) bU[kt] = load_b_frag(Ur_pk, kt, nt, lane);
    float rxv[8];
#pragma unroll
    for (int v = 0; v < 8; ++v)
      rxv[v] = rx[(size_t)(e0 + v + 8 * half) * HDIM + n] + bu;
    float sumg[8];
#pragma unroll
    for (int v = 0; v < 8; ++v) sumg[v] = 0.f;
    for (int nb = 0; nb < 6; ++nb) {
      v8f acc = vzero8();
#pragma unroll
      for (int kt = 0; kt < 8; ++kt)
        acc = wmma_bf16(load_a_frag(hnei, nb * 16, HN_STRIDE, kt * 32, lane),
                        bU[kt], acc);
#pragma unroll
      for (int v = 0; v < 8; ++v) {
        const int M = v + 8 * half;
        const float r = sigmoidf_(acc[v] + rxv[v]);
        const unsigned w = hnei[(nb * 16 + M) * HN_STRIDE + (n >> 1)];
        const float hv = bf2f((n & 1) ? (w >> 16) : (w & 0xFFFFu));
        sumg[v] += r * hv;
      }
    }
    // sum_gated overwrites sum_h (snapshotted above) in xa cols 128..383
    unsigned short* x16 = (unsigned short*)xa;
#pragma unroll
    for (int v = 0; v < 8; ++v) {
      const int M = v + 8 * half;
      x16[M * (XA_STRIDE * 2) + 128 + n] = (unsigned short)f2bf(sumg[v]);
    }
  }
  __syncthreads();

  // Phase 4: pre_h = tanh([fmess|sum_gated] @ W_h + b_h); GRU update + store.
#pragma unroll
  for (int nti = 0; nti < 4; ++nti) {
    const int nt = wave + nti * 4;
    const int n = nt * 16 + mlo;
    v8f acc = vzero8();
    for (int kt = 0; kt < 12; ++kt)
      acc = wmma_bf16(load_a_frag(xa, 0, XA_STRIDE, kt * 32, lane),
                      load_b_frag(Wh_pk, kt, nt, lane), acc);
    const float bhv = b_h[n];
#pragma unroll
    for (int v = 0; v < 8; ++v) {
      const int M = v + 8 * half;
      const float p = tanhf(acc[v] + bhv);
      const float z = z_r[nti][v];
      float hn = (1.f - z) * sh_r[nti][v] + z * p;
      const int edge = e0 + M;
      if (edge == 0) hn = 0.f;               // null-message mask
      if (OUT_F32)
        ((float*)h_out)[(size_t)edge * HDIM + n] = hn;
      else
        ((unsigned short*)h_out)[(size_t)edge * HDIM + n] = (unsigned short)f2bf(hn);
    }
  }
}

// ---- node readout: relu([fnode | sum(h[agraph])] @ W_o + b_o) * mask ----
__global__ __launch_bounds__(128)
void node_out_kernel(const float* __restrict__ fnode, const int* __restrict__ agraph,
                     const float* __restrict__ h, const unsigned* __restrict__ Wo_pk,
                     const float* __restrict__ b_o, const float* __restrict__ mask,
                     float* __restrict__ out) {
  __shared__ unsigned hnei[96 * HN_STRIDE];
  __shared__ unsigned xa[16 * XA_STRIDE];
  __shared__ int aidx[96];
  const int t = threadIdx.x;
  const int v0 = blockIdx.x * 16;
  if (t < 96) {
    int node = v0 + (t & 15);
    if (node >= N_NODES) node = N_NODES - 1;   // tail clamp (results discarded)
    aidx[t] = agraph[(size_t)node * MAX_NB + (t >> 4)];
  }
  __syncthreads();
  const int sub = t & 15, rg = t >> 4;
  for (int r = rg; r < 96; r += 8) {
    const float4* src = (const float4*)(h + (size_t)aidx[r] * HDIM) + sub * 4;
    unsigned* dst = hnei + r * HN_STRIDE + sub * 8;
#pragma unroll
    for (int i = 0; i < 4; ++i) {
      float4 x = src[i];
      dst[2 * i]     = pack2(x.x, x.y);
      dst[2 * i + 1] = pack2(x.z, x.w);
    }
  }
  for (int r = rg; r < 16; r += 8) {
    int node = v0 + r;
    if (node >= N_NODES) node = N_NODES - 1;
    const float4* src = (const float4*)(fnode + (size_t)node * IN_DIM) + sub * 2;
    unsigned* dst = xa + r * XA_STRIDE + sub * 4;
#pragma unroll
    for (int i = 0; i < 2; ++i) {
      float4 x = src[i];
      dst[2 * i]     = pack2(x.x, x.y);
      dst[2 * i + 1] = pack2(x.z, x.w);
    }
  }
  __syncthreads();
  {
    const int e = t >> 3;
    const int w0 = (t & 7) * 16;
#pragma unroll
    for (int i = 0; i < 16; ++i) {
      float s0 = 0.f, s1 = 0.f;
#pragma unroll
      for (int nb = 0; nb < 6; ++nb) {
        const unsigned w = hnei[(nb * 16 + e) * HN_STRIDE + w0 + i];
        s0 += bf2f(w & 0xFFFFu);
        s1 += bf2f(w >> 16);
      }
      xa[e * XA_STRIDE + 64 + w0 + i] = pack2(s0, s1);
    }
  }
  __syncthreads();
  const int lane = t & 31, wave = t >> 5;
  const int mlo = lane & 15, half = lane >> 4;
  for (int nti = 0; nti < 4; ++nti) {
    const int nt = wave + nti * 4;
    const int n = nt * 16 + mlo;
    v8f acc = vzero8();
    for (int kt = 0; kt < 12; ++kt)
      acc = wmma_bf16(load_a_frag(xa, 0, XA_STRIDE, kt * 32, lane),
                      load_b_frag(Wo_pk, kt, nt, lane), acc);
    const float bo = b_o[n];
#pragma unroll
    for (int v = 0; v < 8; ++v) {
      const int node = v0 + v + 8 * half;
      if (node < N_NODES)
        out[(size_t)node * HDIM + n] = fmaxf(acc[v] + bo, 0.f) * mask[node];
    }
  }
}

extern "C" void kernel_launch(void* const* d_in, const int* in_sizes, int n_in,
                              void* d_out, int out_size, void* d_ws, size_t ws_size,
                              hipStream_t stream) {
  const float* fnode  = (const float*)d_in[0];
  const float* fmess  = (const float*)d_in[1];
  const int*   agraph = (const int*)d_in[2];
  const int*   bgraph = (const int*)d_in[3];
  const float* mask   = (const float*)d_in[4];
  const float* W_z    = (const float*)d_in[5];
  const float* b_z    = (const float*)d_in[6];
  const float* W_r    = (const float*)d_in[7];
  const float* U_r    = (const float*)d_in[8];
  const float* b_ur   = (const float*)d_in[9];
  const float* W_h    = (const float*)d_in[10];
  const float* b_h    = (const float*)d_in[11];
  const float* W_o    = (const float*)d_in[12];
  const float* b_o    = (const float*)d_in[13];

  float* out_nodes = (float*)d_out;
  float* h_final   = out_nodes + (size_t)N_NODES * HDIM;   // second output: h (f32)

  char* ws = (char*)d_ws;
  const size_t hbf_bytes = (size_t)N_MSGS * HDIM * 2;       // bf16 hidden state
  unsigned short* hbfA = (unsigned short*)ws; ws += hbf_bytes;
  unsigned short* hbfB = (unsigned short*)ws; ws += hbf_bytes;
  float* rx = (float*)ws;          ws += (size_t)N_MSGS * HDIM * 4;
  unsigned short* fm_bf = (unsigned short*)ws; ws += (size_t)N_MSGS * IN_DIM * 2;
  unsigned* Wr_pk = (unsigned*)ws; ws += (size_t)64  * 256 * 4;
  unsigned* Wz_pk = (unsigned*)ws; ws += (size_t)192 * 256 * 4;
  unsigned* Ur_pk = (unsigned*)ws; ws += (size_t)128 * 256 * 4;
  unsigned* Wh_pk = (unsigned*)ws; ws += (size_t)192 * 256 * 4;
  unsigned* Wo_pk = (unsigned*)ws; ws += (size_t)192 * 256 * 4;

  hipMemsetAsync(hbfA, 0, hbf_bytes, stream);
  pack_weight_kernel<<<64,  256, 0, stream>>>(W_r, Wr_pk);   // K=128
  pack_weight_kernel<<<192, 256, 0, stream>>>(W_z, Wz_pk);   // K=384
  pack_weight_kernel<<<128, 256, 0, stream>>>(U_r, Ur_pk);   // K=256
  pack_weight_kernel<<<192, 256, 0, stream>>>(W_h, Wh_pk);   // K=384
  pack_weight_kernel<<<192, 256, 0, stream>>>(W_o, Wo_pk);   // K=384
  tobf16_kernel<<<2500, 256, 0, stream>>>(fmess, (unsigned*)fm_bf,
                                          N_MSGS * IN_DIM / 2);
  rx_kernel<<<N_MSGS / 16, 128, 0, stream>>>(fmess, Wr_pk, rx);

  const int gs = N_MSGS / 16;   // 5000 blocks, 16 edges each
  // steps 1..4 in bf16 (ping-pong), step 5 writes f32 h into d_out
  mpn_step_kernel<false><<<gs, 128, 0, stream>>>(hbfA, hbfB, fm_bf, bgraph, rx,
                                                 Wz_pk, Ur_pk, Wh_pk, b_z, b_ur, b_h);
  mpn_step_kernel<false><<<gs, 128, 0, stream>>>(hbfB, hbfA, fm_bf, bgraph, rx,
                                                 Wz_pk, Ur_pk, Wh_pk, b_z, b_ur, b_h);
  mpn_step_kernel<false><<<gs, 128, 0, stream>>>(hbfA, hbfB, fm_bf, bgraph, rx,
                                                 Wz_pk, Ur_pk, Wh_pk, b_z, b_ur, b_h);
  mpn_step_kernel<false><<<gs, 128, 0, stream>>>(hbfB, hbfA, fm_bf, bgraph, rx,
                                                 Wz_pk, Ur_pk, Wh_pk, b_z, b_ur, b_h);
  mpn_step_kernel<true><<<gs, 128, 0, stream>>>(hbfA, h_final, fm_bf, bgraph, rx,
                                                Wz_pk, Ur_pk, Wh_pk, b_z, b_ur, b_h);

  node_out_kernel<<<(N_NODES + 15) / 16, 128, 0, stream>>>(fnode, agraph, h_final,
                                                           Wo_pk, b_o, mask, out_nodes);
  (void)in_sizes; (void)n_in; (void)out_size; (void)ws_size;
}
